// CausalSelfAttention_6485400617223
// MI455X (gfx1250) — matrix operand order repaired and checked by reading, not compiled, not run on hardware
//
#include <hip/hip_runtime.h>
#include <hip/hip_bf16.h>

typedef __attribute__((ext_vector_type(16))) __bf16 v16bf;
typedef __attribute__((ext_vector_type(8)))  float  v8f;
typedef int i32x4 __attribute__((__vector_size__(16)));

union Frag {
    v16bf v;
    unsigned int u[8];
    unsigned short s[16];
};

// A-matrix 16x32 bf16 layout (CDNA5 ISA 7.12.2):
// lane = M row (lane&15), lane>>4 selects K-half pattern.
__device__ __forceinline__ int kmapA(int v, int kb) {
    return 2 * v + (v < 4 ? 0 : 8) + 8 * kb;
}

__device__ __forceinline__ unsigned short f2bf(float f) {
    return __builtin_bit_cast(unsigned short, (__bf16)f);
}

#define D_IN   1024
#define D_OUT  1024
#define TOK    8192      // B*T
#define TSEQ   2048

// gfx1250 async global->LDS copy (guarded; sync b128 fallback)
#if defined(__HIP_DEVICE_COMPILE__) && defined(__gfx1250__) && \
    __has_builtin(__builtin_amdgcn_global_load_async_to_lds_b128) && \
    __has_builtin(__builtin_amdgcn_s_wait_asynccnt)
#define USE_ASYNC_LDS 1
#endif

__device__ __forceinline__ void copy16_g2l(const unsigned short* g,
                                           unsigned short* l) {
#ifdef USE_ASYNC_LDS
    __builtin_amdgcn_global_load_async_to_lds_b128(
        (__attribute__((address_space(1))) i32x4*)g,
        (__attribute__((address_space(3))) i32x4*)l, 0, 0);
#else
    *(uint4*)l = *(const uint4*)g;
#endif
}

__device__ __forceinline__ void async_wait_all() {
#ifdef USE_ASYNC_LDS
    __builtin_amdgcn_s_wait_asynccnt(0);
#endif
}

// ---------------------------------------------------------------------------
// Kernel 0: f32 -> bf16 pack (8 elements / thread, b128 in, b128 out)
// ---------------------------------------------------------------------------
__global__ __launch_bounds__(256)
void cvt_bf16_kernel(const float* __restrict__ src,
                     unsigned short* __restrict__ dst, int n8) {
    int i = blockIdx.x * 256 + threadIdx.x;
    if (i >= n8) return;
    const float4* s = (const float4*)src + (size_t)i * 2;
    float4 a = s[0], b = s[1];
    union { unsigned short h[8]; uint4 q; } o;
    o.h[0] = f2bf(a.x); o.h[1] = f2bf(a.y); o.h[2] = f2bf(a.z); o.h[3] = f2bf(a.w);
    o.h[4] = f2bf(b.x); o.h[5] = f2bf(b.y); o.h[6] = f2bf(b.z); o.h[7] = f2bf(b.w);
    ((uint4*)dst)[i] = o.q;
}

// ---------------------------------------------------------------------------
// Kernel 1: QKV projection on bf16.  Block tile 128(M) x 64(N), 8 waves each
// 32x32 (2x2 WMMA frags), K step 32.  Tiles staged via async b128 -> LDS.
// z==0 -> Q [tok][e]; z==1 -> Kt [b][e][t] (transposed); z==2 -> V [tok][e].
// Output tile bounced through LDS for coalesced b128 stores.
// ---------------------------------------------------------------------------
#define XS_STRIDE 40   // shorts: 80B rows, 16B aligned, conflict-free frag reads
#define WS_STRIDE 72   // shorts: 144B rows, 16B aligned

__global__ __launch_bounds__(256)
void qkv_gemm_kernel(const unsigned short* __restrict__ xb,
                     const unsigned short* __restrict__ Wb,   // [3][1024][1024]
                     unsigned short* __restrict__ Qb,
                     unsigned short* __restrict__ Kt,
                     unsigned short* __restrict__ Vb) {
    __shared__ __align__(16) unsigned short Xs[128 * XS_STRIDE];
    __shared__ __align__(16) unsigned short Wsh[32 * WS_STRIDE];
    __shared__ __align__(16) unsigned short Ct[128 * 64];

    const int tid  = threadIdx.x;
    const int lane = tid & 31;
    const int wid  = tid >> 5;
    const int wm   = wid & 3;
    const int wn   = wid >> 2;
    const int lm   = lane & 15;
    const int kb   = lane >> 4;

    const int z       = blockIdx.z;
    const int rowBase = blockIdx.x * 128;   // token base
    const int colBase = blockIdx.y * 64;    // feature base
    const unsigned short* W = Wb + (size_t)z * D_IN * D_OUT;

    v8f acc[2][2];
    #pragma unroll
    for (int i = 0; i < 2; ++i)
        #pragma unroll
        for (int j = 0; j < 2; ++j)
            acc[i][j] = {};

    for (int kk = 0; kk < D_IN; kk += 32) {
        // X tile 128x32 bf16: 512 x 16B chunks, 2 per thread
        #pragma unroll
        for (int i = 0; i < 2; ++i) {
            int c = tid + i * 256;
            int row = c >> 2, part = c & 3;
            copy16_g2l(xb + (size_t)(rowBase + row) * D_IN + kk + part * 8,
                       &Xs[row * XS_STRIDE + part * 8]);
        }
        // W tile 32x64 bf16: 256 x 16B chunks, 1 per thread
        {
            int row = tid >> 3, part = tid & 7;
            copy16_g2l(W + (size_t)(kk + row) * D_OUT + colBase + part * 8,
                       &Wsh[row * WS_STRIDE + part * 8]);
        }
        async_wait_all();
        __syncthreads();

        Frag a[2], b[2];
        #pragma unroll
        for (int mi = 0; mi < 2; ++mi) {
            int row = wm * 32 + mi * 16 + lm;
            #pragma unroll
            for (int v = 0; v < 8; ++v)
                a[mi].u[v] =
                    *(const unsigned int*)&Xs[row * XS_STRIDE + kmapA(v, kb)];
        }
        #pragma unroll
        for (int ni = 0; ni < 2; ++ni) {
            int nb = wn * 32 + ni * 16;      // B: lane = N column, elements over K (table order)
            #pragma unroll
            for (int s = 0; s < 16; ++s)
                b[ni].s[s] =
                    Wsh[(kmapA(s >> 1, kb) + (s & 1)) * WS_STRIDE + nb + lm];
        }
        #pragma unroll
        for (int mi = 0; mi < 2; ++mi)
            #pragma unroll
            for (int ni = 0; ni < 2; ++ni)
                acc[mi][ni] = __builtin_amdgcn_wmma_f32_16x16x32_bf16(
                    false, a[mi].v, false, b[ni].v, (short)0, acc[mi][ni],
                    false, false);
        __syncthreads();
    }

    // write accumulators to LDS tile (feature-major when transposing for Kt)
    #pragma unroll
    for (int mi = 0; mi < 2; ++mi) {
        #pragma unroll
        for (int ni = 0; ni < 2; ++ni) {
            #pragma unroll
            for (int r = 0; r < 8; ++r) {
                int ml = wm * 32 + mi * 16 + r + 8 * kb;   // 0..127 (token)
                int nl = wn * 32 + ni * 16 + lm;           // 0..63  (feature)
                unsigned short hv = f2bf(acc[mi][ni][r]);
                if (z == 1) Ct[nl * 128 + ml] = hv;
                else        Ct[ml * 64 + nl]  = hv;
            }
        }
    }
    __syncthreads();

    // coalesced b128 writeback
    if (z == 1) {
        const int bbb = rowBase / TSEQ;
        const int t0  = rowBase % TSEQ;     // 128-row tile never spans batches
        #pragma unroll
        for (int i = 0; i < 4; ++i) {
            int c = tid + i * 256;          // 1024 chunks: 64 rows x 16
            int nrow = c >> 4, part = c & 15;
            *(uint4*)(Kt + (size_t)(bbb * D_OUT + colBase + nrow) * TSEQ +
                      t0 + part * 8) = *(const uint4*)&Ct[nrow * 128 + part * 8];
        }
    } else {
        unsigned short* dst = (z == 0) ? Qb : Vb;
        #pragma unroll
        for (int i = 0; i < 4; ++i) {
            int c = tid + i * 256;          // 1024 chunks: 128 rows x 8
            int mrow = c >> 3, part = c & 7;
            *(uint4*)(dst + (size_t)(rowBase + mrow) * D_OUT + colBase +
                      part * 8) = *(const uint4*)&Ct[mrow * 64 + part * 8];
        }
    }
}

// ---------------------------------------------------------------------------
// Kernel 2: flash attention.  One block = 16 query rows; 8 waves split d_out
// into 128-wide slices.  Key tiles of 32, online softmax, causal.
// ---------------------------------------------------------------------------
__global__ __launch_bounds__(256)
void flash_attn_kernel(const unsigned short* __restrict__ Qb,
                       const unsigned short* __restrict__ Kt,
                       const unsigned short* __restrict__ Vb,
                       float* __restrict__ out) {
    __shared__ __align__(16) float  Sl[16 * 33];    // score tile 16x32 (padded)
    __shared__ __align__(16) __bf16 Pl[16 * 34];    // probs bf16 (padded)
    __shared__ float fac[16];

    const int tid  = threadIdx.x;
    const int lane = tid & 31;
    const int wid  = tid >> 5;
    const int lm   = lane & 15;
    const int half = lane >> 4;

    const int q0g = blockIdx.x * 16;        // global token base
    const int bb  = q0g / TSEQ;
    const int q0  = q0g % TSEQ;             // local (in-batch) query base

    // resident Q fragments: 4 chunks of 32 dims covering this wave's 128 dims
    Frag qf[4];
    #pragma unroll
    for (int c = 0; c < 4; ++c) {
        int dq = wid * 128 + c * 32;
        #pragma unroll
        for (int v = 0; v < 8; ++v)
            qf[c].u[v] = *(const unsigned int*)
                (Qb + (size_t)(q0g + lm) * D_OUT + dq + kmapA(v, half));
    }

    v8f o[8];
    #pragma unroll
    for (int f = 0; f < 8; ++f) o[f] = {};

    float mrow = -INFINITY, lrow = 0.0f;    // meaningful for tid<16

    const int nkt = (q0 + 15) / 32 + 1;     // causal key-tile bound
    for (int kt = 0; kt < nkt; ++kt) {
        const int k0 = kt * 32;

        __syncthreads();
        for (int i = tid; i < 16 * 33; i += 256) Sl[i] = 0.0f;
        __syncthreads();

        // partial scores over this wave's 128 dims: S(16x32) = Q * Kt
        v8f s0 = {}, s1 = {};
        #pragma unroll
        for (int c = 0; c < 4; ++c) {
            int d = wid * 128 + c * 32;                     // B: lane = key column
            const unsigned short* kbase =
                Kt + ((size_t)(bb * D_OUT + d)) * TSEQ + k0 + lm;
            Frag b0, b1;
            #pragma unroll
            for (int s = 0; s < 16; ++s) {
                size_t kd = (size_t)(kmapA(s >> 1, half) + (s & 1)) * TSEQ;
                b0.s[s] = kbase[kd];
                b1.s[s] = kbase[kd + 16];
            }
            s0 = __builtin_amdgcn_wmma_f32_16x16x32_bf16(
                false, qf[c].v, false, b0.v, (short)0, s0, false, false);
            s1 = __builtin_amdgcn_wmma_f32_16x16x32_bf16(
                false, qf[c].v, false, b1.v, (short)0, s1, false, false);
        }
        // cross-wave reduction into LDS (ds_add_f32)
        #pragma unroll
        for (int r = 0; r < 8; ++r) {
            int m = r + 8 * half;
            atomicAdd(&Sl[m * 33 + lm],      s0[r]);
            atomicAdd(&Sl[m * 33 + 16 + lm], s1[r]);
        }
        __syncthreads();

        // online softmax row update (lanes 0..15 of wave 0 own one row each)
        if (tid < 16) {
            const int m = tid;
            float sv[32], mt = -INFINITY;
            #pragma unroll
            for (int j = 0; j < 32; ++j) {
                int kg = k0 + j;
                float s = (kg <= q0 + m) ? Sl[m * 33 + j] * 0.03125f
                                         : -INFINITY;      // 1/sqrt(1024)
                sv[j] = s;
                mt = fmaxf(mt, s);
            }
            float mnew = fmaxf(mrow, mt);
            float f = __expf(mrow - mnew);
            float psum = 0.0f;
            #pragma unroll
            for (int j = 0; j < 32; ++j) {
                float p = __expf(sv[j] - mnew);
                psum += p;
                Pl[m * 34 + j] = (__bf16)p;
            }
            lrow = lrow * f + psum;
            mrow = mnew;
            fac[m] = f;
        }
        __syncthreads();

        // rescale O, then O += P * V
        float fr[8];
        #pragma unroll
        for (int r = 0; r < 8; ++r) fr[r] = fac[r + 8 * half];
        #pragma unroll
        for (int f = 0; f < 8; ++f)
            #pragma unroll
            for (int r = 0; r < 8; ++r)
                o[f][r] *= fr[r];

        Frag pa;                                           // P in A-layout
        #pragma unroll
        for (int v = 0; v < 8; ++v)
            pa.u[v] = *(const unsigned int*)&Pl[lm * 34 + kmapA(v, half)];

        const unsigned short* vrow =
            Vb + (size_t)(bb * TSEQ + k0) * D_OUT + lm;    // B: lane = feature column
        #pragma unroll
        for (int f = 0; f < 8; ++f) {
            int fb = wid * 128 + f * 16;
            Frag vf;
            #pragma unroll
            for (int s = 0; s < 16; ++s)
                vf.s[s] = vrow[(size_t)(kmapA(s >> 1, half) + (s & 1)) * D_OUT + fb];
            o[f] = __builtin_amdgcn_wmma_f32_16x16x32_bf16(
                false, pa.v, false, vf.v, (short)0, o[f], false, false);
        }
    }

    __syncthreads();
    if (tid < 16) fac[tid] = 1.0f / lrow;
    __syncthreads();

    float inv[8];
    #pragma unroll
    for (int r = 0; r < 8; ++r) inv[r] = fac[r + 8 * half];
    #pragma unroll
    for (int f = 0; f < 8; ++f) {
        int fb = wid * 128 + f * 16;
        #pragma unroll
        for (int r = 0; r < 8; ++r) {
            int row = r + 8 * half;
            out[(size_t)(q0g + row) * D_OUT + fb + lm] = o[f][r] * inv[r];
        }
    }
}

// ---------------------------------------------------------------------------
extern "C" void kernel_launch(void* const* d_in, const int* in_sizes, int n_in,
                              void* d_out, int out_size, void* d_ws, size_t ws_size,
                              hipStream_t stream) {
    const float* x  = (const float*)d_in[0];
    const float* Wq = (const float*)d_in[1];
    const float* Wk = (const float*)d_in[2];
    const float* Wv = (const float*)d_in[3];
    float* out = (float*)d_out;

    unsigned short* xb = (unsigned short*)d_ws;              // 16 MB
    unsigned short* Wb = xb + (size_t)TOK * D_IN;            //  6 MB (3x)
    unsigned short* Qb = Wb + (size_t)3 * D_IN * D_OUT;      // 16 MB
    unsigned short* Kt = Qb + (size_t)TOK * D_OUT;           // 16 MB
    unsigned short* Vb = Kt + (size_t)TOK * D_OUT;           // 16 MB

    const int nx8 = TOK * D_IN / 8;                          // 1048576
    const int nw8 = D_IN * D_OUT / 8;                        // 131072
    cvt_bf16_kernel<<<nx8 / 256, 256, 0, stream>>>(x, xb, nx8);
    cvt_bf16_kernel<<<nw8 / 256, 256, 0, stream>>>(Wq, Wb, nw8);
    cvt_bf16_kernel<<<nw8 / 256, 256, 0, stream>>>(Wk, Wb + (size_t)D_IN * D_OUT, nw8);
    cvt_bf16_kernel<<<nw8 / 256, 256, 0, stream>>>(Wv, Wb + (size_t)2 * D_IN * D_OUT, nw8);

    qkv_gemm_kernel<<<dim3(TOK / 128, D_OUT / 64, 3), 256, 0, stream>>>(
        xb, Wb, Qb, Kt, Vb);
    flash_attn_kernel<<<dim3(TOK / 16), 256, 0, stream>>>(Qb, Kt, Vb, out);
}
